// MultiHeadAttention_17197049053932
// MI455X (gfx1250) — compile-verified, matrix operand order checked by reading, not run on hardware
//
#include <hip/hip_runtime.h>
#include <hip/hip_bf16.h>

#define NROW 8192
#define NCOL 8192

typedef __attribute__((ext_vector_type(16))) __bf16 v16bf;
typedef __attribute__((ext_vector_type(8)))  __bf16 v8bf;
typedef __attribute__((ext_vector_type(8)))  float  v8f;
typedef __attribute__((ext_vector_type(4)))  float  v4f;

__device__ __forceinline__ v16bf bf_cat(v8bf lo, v8bf hi) {
  v16bf r;
#pragma unroll
  for (int i = 0; i < 8; ++i) { r[i] = lo[i]; r[i + 8] = hi[i]; }
  return r;
}

__device__ __forceinline__ v8f wmma_bf16(v16bf a, v16bf b, v8f c) {
  return __builtin_amdgcn_wmma_f32_16x16x32_bf16(false, a, false, b, (short)0, c, false, false);
}

// ---------------------------------------------------------------------------
// k0: transpose-convert the four 256x256 weight matrices to bf16:
//     wt[m][n*256+k] = (bf16) W_m[k*256+n]   (so B-fragments are contiguous)
// ---------------------------------------------------------------------------
__global__ void k0_wt(const float* __restrict__ Wq, const float* __restrict__ Wk,
                      const float* __restrict__ Wv, const float* __restrict__ Wo,
                      __bf16* __restrict__ wt) {
  const int m = blockIdx.y;
  const int k = blockIdx.x;
  const int n = threadIdx.x;
  const float* W = (m == 0) ? Wq : (m == 1) ? Wk : (m == 2) ? Wv : Wo;
  wt[m * 65536 + n * 256 + k] = (__bf16)W[k * 256 + n];
}

// ---------------------------------------------------------------------------
// k1: projections via bf16 WMMA (one wave = 16 rows x 256 cols, K=256).
//   tgt 0: xq = x1@Wq+bq (row-major bf16); tgt 1: xk = x2@Wk+bk (row-major);
//   tgt 2: xvT = (x2@Wv+bv)^T  (bf16 [256,8192], packed b128 stores)
// ---------------------------------------------------------------------------
__global__ void __launch_bounds__(128) k1_proj(
    const float* __restrict__ x1, const float* __restrict__ x2,
    const float* __restrict__ bq, const float* __restrict__ bk,
    const float* __restrict__ bv, const __bf16* __restrict__ wt,
    __bf16* __restrict__ xq, __bf16* __restrict__ xk, __bf16* __restrict__ xvT) {
  const int lane  = threadIdx.x & 31;
  const int w     = threadIdx.x >> 5;
  const int ln    = lane & 15;
  const int h     = lane >> 4;
  const int tgt   = blockIdx.y;
  const int ibase = (blockIdx.x * 4 + w) * 16;

  const float*  A    = (tgt == 0) ? x1 : x2;
  const float*  bias = (tgt == 0) ? bq : (tgt == 1) ? bk : bv;
  const __bf16* Wt   = wt + tgt * 65536;

  const float* arow = A + (size_t)(ibase + ln) * 256;
  v16bf aA[8];
#pragma unroll
  for (int kc = 0; kc < 8; ++kc) {
    v4f a0 = *(const v4f*)(arow + kc * 32 + h * 8);
    v4f a1 = *(const v4f*)(arow + kc * 32 + h * 8 + 4);
    v4f a2 = *(const v4f*)(arow + kc * 32 + 16 + h * 8);
    v4f a3 = *(const v4f*)(arow + kc * 32 + 16 + h * 8 + 4);
    v16bf af;
#pragma unroll
    for (int i = 0; i < 4; ++i) {
      af[i]      = (__bf16)a0[i];
      af[4 + i]  = (__bf16)a1[i];
      af[8 + i]  = (__bf16)a2[i];
      af[12 + i] = (__bf16)a3[i];
    }
    aA[kc] = af;
  }

#pragma unroll 1
  for (int nt = 0; nt < 16; ++nt) {
    const __bf16* wrow = Wt + (size_t)(nt * 16 + ln) * 256;
    v16bf bw[8];
#pragma unroll
    for (int kc = 0; kc < 8; ++kc)
      bw[kc] = bf_cat(*(const v8bf*)(wrow + kc * 32 + h * 8),
                      *(const v8bf*)(wrow + kc * 32 + 16 + h * 8));
    float bvv = bias[nt * 16 + ln];
    v8f c;
#pragma unroll
    for (int g = 0; g < 8; ++g) c[g] = bvv;
#pragma unroll
    for (int kc = 0; kc < 8; ++kc) c = wmma_bf16(aA[kc], bw[kc], c);

    if (tgt < 2) {
      __bf16* o = (tgt == 0) ? xq : xk;
#pragma unroll
      for (int g = 0; g < 8; ++g)
        o[(size_t)(ibase + g + 8 * h) * 256 + nt * 16 + ln] = (__bf16)c[g];
    } else {
      v8bf pk;
#pragma unroll
      for (int g = 0; g < 8; ++g) pk[g] = (__bf16)c[g];
      *(v8bf*)(xvT + (size_t)(nt * 16 + ln) * NCOL + ibase + 8 * h) = pk;
    }
  }
}

// ---------------------------------------------------------------------------
// k2: fused flash-style SupCon attention, cooperative-LDS version.
// Block = 4 waves = 64 q rows. Per 128-col j tile the block stages the xk tile
// (64 KB, contiguous) and xvT tile (64 KB) into LDS; each wave then runs:
//   scores (bf16 WMMA from LDS) -> online softmax + masked P -> LDS ->
//   acc += P @ V (bf16 WMMA from LDS).
// All loops that index register arrays are FULLY unrolled (no dynamic
// register-array indices -> no alloca/scratch spills).
// Dynamic LDS: XK 32768 + XV 32768 + 4x4096 P/res = 81920 bf16 = 160 KB.
// ---------------------------------------------------------------------------
__global__ void __launch_bounds__(128) k2_flash(
    const __bf16* __restrict__ xq, const __bf16* __restrict__ xk,
    const __bf16* __restrict__ xvT, const int* __restrict__ ly1,
    const int* __restrict__ ly2, const float* __restrict__ x1,
    const __bf16* __restrict__ WoT, const float* __restrict__ bout,
    float* __restrict__ d_out, float* __restrict__ bsum) {
  extern __shared__ __bf16 smem[];
  __bf16* XK = smem;              // [128][256]
  __bf16* XV = smem + 32768;      // [256][128]
  const int tid   = threadIdx.x;
  const int lane  = tid & 31;
  const int w     = tid >> 5;
  const int ln    = lane & 15;
  const int h     = lane >> 4;
  const int ibase = (blockIdx.x * 4 + w) * 16;
  __bf16* Pb = smem + 65536 + w * 4096;   // per-wave P tile / res tile

  int l1r[8];
#pragma unroll
  for (int g = 0; g < 8; ++g) l1r[g] = ly1[ibase + g + 8 * h];

  // resident xq A-fragments (loaded once)
  const __bf16* qrow = xq + (size_t)(ibase + ln) * 256;
  v16bf aq[8];
#pragma unroll
  for (int kc = 0; kc < 8; ++kc)
    aq[kc] = bf_cat(*(const v8bf*)(qrow + kc * 32 + h * 8),
                    *(const v8bf*)(qrow + kc * 32 + 16 + h * 8));

  float Mrow[8], zp[8], sp[8], pp[8];
#pragma unroll
  for (int g = 0; g < 8; ++g) { Mrow[g] = -3.0e38f; zp[g] = 0.f; sp[g] = 0.f; pp[g] = 0.f; }

  v8f acc[16];
#pragma unroll
  for (int t = 0; t < 16; ++t)
#pragma unroll
    for (int g = 0; g < 8; ++g) acc[t][g] = 0.f;

  for (int jt = 0; jt < NCOL; jt += 128) {
    // ---- cooperative staging of xk / xvT tiles ----
    __syncthreads();                       // previous tile fully consumed
    {
      const v8bf* srcK = (const v8bf*)(xk + (size_t)jt * 256);
      v8bf* dstK = (v8bf*)XK;
#pragma unroll 1
      for (int i = tid; i < 4096; i += 128) dstK[i] = srcK[i];
      v8bf* dstV = (v8bf*)XV;
#pragma unroll 1
      for (int i = tid; i < 4096; i += 128) {
        int vrow = i >> 4;
        int cc   = i & 15;
        dstV[i] = *(const v8bf*)(xvT + (size_t)vrow * NCOL + jt + cc * 8);
      }
      if (jt + 128 < NCOL) {               // L2 prefetch of next tile
        __builtin_prefetch(xk + (size_t)(jt + 128) * 256 + tid * 256, 0, 1);
        __builtin_prefetch(xvT + (size_t)(tid * 2) * NCOL + jt + 128, 0, 1);
      }
    }
    __syncthreads();

    // ---- scores: 16x128 tile, K=256 (64 WMMA, operands from LDS) ----
    v8f C[8];
#pragma unroll
    for (int f = 0; f < 8; ++f)
#pragma unroll
      for (int g = 0; g < 8; ++g) C[f][g] = 0.f;

#pragma unroll
    for (int kc = 0; kc < 8; ++kc) {       // FULL unroll: aq[kc] stays in VGPRs
      v16bf bk[8];
#pragma unroll
      for (int f = 0; f < 8; ++f) {
        const __bf16* krow = XK + (16 * f + ln) * 256;
        bk[f] = bf_cat(*(const v8bf*)(krow + kc * 32 + h * 8),
                       *(const v8bf*)(krow + kc * 32 + 16 + h * 8));
      }
#pragma unroll
      for (int f = 0; f < 8; ++f) C[f] = wmma_bf16(aq[kc], bk[f], C[f]);
    }
#pragma unroll
    for (int f = 0; f < 8; ++f)
#pragma unroll
      for (int g = 0; g < 8; ++g) C[f][g] *= 0.0625f;   // 1/sqrt(256)

    // ---- online softmax: tile row max -> rescale running state ----
    float pm[8];
#pragma unroll
    for (int g = 0; g < 8; ++g) pm[g] = -3.0e38f;
#pragma unroll
    for (int f = 0; f < 8; ++f)
#pragma unroll
      for (int g = 0; g < 8; ++g) pm[g] = fmaxf(pm[g], C[f][g]);
#pragma unroll
    for (int mk = 1; mk < 16; mk <<= 1)
#pragma unroll
      for (int g = 0; g < 8; ++g) pm[g] = fmaxf(pm[g], __shfl_xor(pm[g], mk, 32));

    float rr[8];
#pragma unroll
    for (int g = 0; g < 8; ++g) {
      float nm = fmaxf(Mrow[g], pm[g]);
      rr[g]    = __expf(Mrow[g] - nm);
      Mrow[g]  = nm;
      zp[g]   *= rr[g];
    }
#pragma unroll
    for (int t = 0; t < 16; ++t)
#pragma unroll
      for (int g = 0; g < 8; ++g) acc[t][g] *= rr[g];

    // ---- exp, stats, masked P -> LDS (row-major 16x128, per-wave) ----
#pragma unroll
    for (int f = 0; f < 8; ++f) {
      int l2 = ly2[jt + 16 * f + ln];
#pragma unroll
      for (int g = 0; g < 8; ++g) {
        float sv = C[f][g];
        float e  = __expf(sv - Mrow[g]);
        zp[g] += e;
        float pb = 0.f;
        if (l1r[g] == l2) { sp[g] += sv; pp[g] += 1.f; pb = e; }
        Pb[(g + 8 * h) * 128 + 16 * f + ln] = (__bf16)pb;
      }
    }

    // ---- acc += P(16x128) @ V(128x256)  (64 WMMA, operands from LDS) ----
#pragma unroll
    for (int kc = 0; kc < 4; ++kc) {       // FULL unroll
      v16bf ap = bf_cat(*(const v8bf*)(Pb + ln * 128 + kc * 32 + h * 8),
                        *(const v8bf*)(Pb + ln * 128 + kc * 32 + 16 + h * 8));
#pragma unroll
      for (int th = 0; th < 2; ++th) {     // FULL unroll: acc indices constant
        v16bf bv[8];
#pragma unroll
        for (int t = 0; t < 8; ++t) {
          const __bf16* vrow = XV + ((th * 8 + t) * 16 + ln) * 128;
          bv[t] = bf_cat(*(const v8bf*)(vrow + kc * 32 + h * 8),
                         *(const v8bf*)(vrow + kc * 32 + 16 + h * 8));
        }
#pragma unroll
        for (int t = 0; t < 8; ++t)
          acc[th * 8 + t] = wmma_bf16(ap, bv[t], acc[th * 8 + t]);
      }
    }
  }

  // ---- final stats: reduce over the 16 lanes of each half ----
#pragma unroll
  for (int mk = 1; mk < 16; mk <<= 1)
#pragma unroll
    for (int g = 0; g < 8; ++g) {
      zp[g] += __shfl_xor(zp[g], mk, 32);
      sp[g] += __shfl_xor(sp[g], mk, 32);
      pp[g] += __shfl_xor(pp[g], mk, 32);
    }
  float Zg[8];
  float mll = 0.f;
#pragma unroll
  for (int g = 0; g < 8; ++g) {
    Zg[g]    = zp[g];
    float lz = __logf(zp[g]);
    float dn = fmaxf(pp[g], 1.f);
    mll += (sp[g] - pp[g] * (Mrow[g] + lz)) / dn;
  }
  mll += __shfl_xor(mll, 16, 32);
  if (lane == 0) bsum[blockIdx.x * 4 + w] = mll;

  // ---- out = acc/Z ; res = x1 + out ; row L2 normalize ----
  float ss[8];
#pragma unroll
  for (int g = 0; g < 8; ++g) ss[g] = 0.f;
#pragma unroll
  for (int t = 0; t < 16; ++t)
#pragma unroll
    for (int g = 0; g < 8; ++g) {
      float o  = acc[t][g] / Zg[g];
      float rv = x1[(size_t)(ibase + g + 8 * h) * 256 + t * 16 + ln] + o;
      acc[t][g] = rv;
      ss[g] += rv * rv;
    }
#pragma unroll
  for (int mk = 1; mk < 16; mk <<= 1)
#pragma unroll
    for (int g = 0; g < 8; ++g) ss[g] += __shfl_xor(ss[g], mk, 32);
  float riv[8];
#pragma unroll
  for (int g = 0; g < 8; ++g) riv[g] = 1.f / fmaxf(sqrtf(ss[g]), 1e-12f);

  // res tile (bf16) to per-wave LDS region (P is dead)
#pragma unroll
  for (int t = 0; t < 16; ++t)
#pragma unroll
    for (int g = 0; g < 8; ++g)
      Pb[(g + 8 * h) * 256 + t * 16 + ln] = (__bf16)(acc[t][g] * riv[g]);

  // ---- output = relu(res @ Wout + bout) ----
  v16bf ra[8];
#pragma unroll
  for (int kc = 0; kc < 8; ++kc)
    ra[kc] = bf_cat(*(const v8bf*)(Pb + ln * 256 + kc * 32 + h * 8),
                    *(const v8bf*)(Pb + ln * 256 + kc * 32 + 16 + h * 8));
#pragma unroll 1
  for (int nt = 0; nt < 16; ++nt) {
    const __bf16* wrow = WoT + (size_t)(nt * 16 + ln) * 256;
    v16bf bw[8];
#pragma unroll
    for (int kc = 0; kc < 8; ++kc)
      bw[kc] = bf_cat(*(const v8bf*)(wrow + kc * 32 + h * 8),
                      *(const v8bf*)(wrow + kc * 32 + 16 + h * 8));
    float bvv = bout[nt * 16 + ln];
    v8f c;
#pragma unroll
    for (int g = 0; g < 8; ++g) c[g] = bvv;
#pragma unroll
    for (int kc = 0; kc < 8; ++kc) c = wmma_bf16(ra[kc], bw[kc], c);
#pragma unroll
    for (int g = 0; g < 8; ++g)
      d_out[1 + (size_t)(ibase + g + 8 * h) * 256 + nt * 16 + ln] = fmaxf(c[g], 0.f);
  }
}

// ---------------------------------------------------------------------------
// k3: deterministic tree reduction of 512 per-wave loss partials.
// ---------------------------------------------------------------------------
__global__ void k3_loss(const float* __restrict__ bs, float* __restrict__ out) {
  __shared__ float sm[256];
  int t = threadIdx.x;
  sm[t] = bs[t] + bs[t + 256];
  __syncthreads();
  for (int s = 128; s > 0; s >>= 1) {
    if (t < s) sm[t] += sm[t + s];
    __syncthreads();
  }
  if (t == 0) out[0] = -sm[0] * (1.0f / 8192.0f);
}

extern "C" void kernel_launch(void* const* d_in, const int* in_sizes, int n_in,
                              void* d_out, int out_size, void* d_ws, size_t ws_size,
                              hipStream_t stream) {
  const float* x1   = (const float*)d_in[0];
  const float* x2   = (const float*)d_in[1];
  const int*   ly1  = (const int*)d_in[2];
  const int*   ly2  = (const int*)d_in[3];
  const float* Wxq  = (const float*)d_in[4];
  const float* bxq  = (const float*)d_in[5];
  const float* Wxk  = (const float*)d_in[6];
  const float* bxk  = (const float*)d_in[7];
  const float* Wxv  = (const float*)d_in[8];
  const float* bxv  = (const float*)d_in[9];
  const float* Wout = (const float*)d_in[10];
  const float* bout = (const float*)d_in[11];
  float* out = (float*)d_out;

  char* ws = (char*)d_ws;
  __bf16* xq   = (__bf16*)(ws);                 // 8192*256 bf16 = 4 MB
  __bf16* xk   = (__bf16*)(ws + 4194304);       // 4 MB
  __bf16* xvT  = (__bf16*)(ws + 8388608);       // 4 MB (transposed [256][8192])
  __bf16* wt   = (__bf16*)(ws + 12582912);      // 4 x 256x256 bf16 = 512 KB
  float*  bsum = (float*)(ws + 13107200);       // 512 floats

  hipLaunchKernelGGL(k0_wt, dim3(256, 4), dim3(256), 0, stream, Wxq, Wxk, Wxv, Wout, wt);
  hipLaunchKernelGGL(k1_proj, dim3(128, 3), dim3(128), 0, stream,
                     x1, x2, bxq, bxk, bxv, wt, xq, xk, xvT);
  hipLaunchKernelGGL(k2_flash, dim3(128), dim3(128), 163840, stream,
                     xq, xk, xvT, ly1, ly2, x1, wt + 3 * 65536, bout, out, bsum);
  hipLaunchKernelGGL(k3_loss, dim3(1), dim3(256), 0, stream, bsum, out);
}
